// LocalLSTM_13657996001377
// MI455X (gfx1250) — matrix operand order repaired and checked by reading, not compiled, not run on hardware
//
#include <hip/hip_runtime.h>

// ---------------------------------------------------------------------------
// MI455X (gfx1250) persistent-state LSTM seq2seq.
// 2 cooperating wave32s = 16 batch rows (split over gate tiles).
// All GEMMs via v_wmma_f32_16x16x32_bf16. Weights pre-packed into WMMA
// B-fragment layout, inputs pre-packed into A-fragment layout.
// h state ping-pong buffered in LDS (fewer barriers); c state in-place.
// ---------------------------------------------------------------------------

typedef __bf16 bf16;
typedef __attribute__((ext_vector_type(16))) __bf16 v16bf;
typedef __attribute__((ext_vector_type(8)))  __bf16 v8bf;
typedef __attribute__((ext_vector_type(8)))  float  v8f;

#define DEVINL __device__ __forceinline__

constexpr int N_TOT  = 10000;
constexpr int T_ALL  = 64;          // T_CTX + HORIZON
constexpr int T_CTX  = 24;
constexpr int HORIZ  = 40;
constexpr int NGROUP = N_TOT / 16;  // 625 (exact)
constexpr int SW     = 152;         // bf16 LDS row stride: 304B = 19*16, bank-friendly
constexpr int SWC    = 132;         // f32  LDS row stride for c-state

// ---- workspace layout (elements of bf16 from d_ws base) -------------------
constexpr size_t FRAG = 512;  // one B fragment: 32 lanes x 16 bf16 = 1KB
constexpr size_t OFF_INP    = 0;
constexpr size_t INP_ELEMS  = (size_t)NGROUP * T_ALL * 512;
constexpr size_t OFF_ENC_IN = OFF_INP + INP_ELEMS;          //   8 frags
constexpr size_t OFF_WIH0   = OFF_ENC_IN +   8 * FRAG;      // 128
constexpr size_t OFF_WHH0   = OFF_WIH0   + 128 * FRAG;      // 128
constexpr size_t OFF_WIH1   = OFF_WHH0   + 128 * FRAG;      // 128
constexpr size_t OFF_WHH1   = OFF_WIH1   + 128 * FRAG;      // 128
constexpr size_t OFF_DEC_IN = OFF_WHH1   + 128 * FRAG;      //   8
constexpr size_t OFF_INWIH  = OFF_DEC_IN +   8 * FRAG;      // 256
constexpr size_t OFF_INWHH  = OFF_INWIH  + 256 * FRAG;      // 128
constexpr size_t OFF_DWIH   = OFF_INWHH  + 128 * FRAG;      // 128
constexpr size_t OFF_DWHH   = OFF_DWIH   + 128 * FRAG;      // 128
constexpr size_t OFF_M1     = OFF_DWHH   + 128 * FRAG;      //  40
constexpr size_t OFF_M2     = OFF_M1     +  40 * FRAG;      //   4
constexpr size_t OFF_END    = OFF_M2     +   4 * FRAG;      // bias f32 follows

// ---------------------------------------------------------------------------
// Prep kernel: pack weight matrix W (Nr x Kc, row-major fp32, gates = act@W.T)
// into bf16 B-fragments. Fragment f = nt*ktC + kt; lane l element e holds
// B[K][N] = W[N][K] with N = nt*16 + (l&15), K = kt*32 + (l>=16?8:0) + e + (e>=8?8:0).
// ---------------------------------------------------------------------------
__global__ void pack_wt(const float* __restrict__ W, bf16* __restrict__ dst,
                        int Nr, int Kc, int ntC, int ktC)
{
    int tid = blockIdx.x * blockDim.x + threadIdx.x;
    int total = ntC * ktC * 32;
    if (tid >= total) return;
    int lane = tid & 31;
    int f    = tid >> 5;
    int kt   = f % ktC, nt = f / ktC;
    int n     = nt * 16 + (lane & 15);
    int kbase = kt * 32 + ((lane >> 4) << 3);
    bf16* o = dst + (size_t)f * FRAG + lane * 16;
    #pragma unroll
    for (int e = 0; e < 16; ++e) {
        int k = kbase + e + ((e >> 3) << 3);
        float v = (n < Nr && k < Kc) ? W[(size_t)n * Kc + k] : 0.0f;
        o[e] = (bf16)v;
    }
}

// ---------------------------------------------------------------------------
// Prep kernel: pack per-timestep model inputs into A-fragment layout.
// A 16x32 bf16 layout: lane l, element e (j=e>>1):
//   M = l&15,  K = (j>=4?16:0) + (l>=16?8:0) + ((j&3)<<1) + (e&1)
// features: 0=x[n,t], 1..2=coords, 3..19=env[n,f,t], 20=areas, 21..31=0
// ---------------------------------------------------------------------------
__global__ void pack_inp(const float* __restrict__ x, const float* __restrict__ coords,
                         const float* __restrict__ env, const float* __restrict__ areas,
                         bf16* __restrict__ dst)
{
    int tid = blockIdx.x * blockDim.x + threadIdx.x;
    int total = NGROUP * T_ALL * 32;
    if (tid >= total) return;
    int lane = tid & 31;
    int gt   = tid >> 5;
    int t = gt % T_ALL;
    int g = gt / T_ALL;
    int row = g * 16 + (lane & 15);
    float ar = areas[row];
    float cx = coords[(size_t)row * 2 + 0];
    float cy = coords[(size_t)row * 2 + 1];
    bf16* o = dst + (size_t)gt * 512 + lane * 16;
    #pragma unroll
    for (int e = 0; e < 16; ++e) {
        int j = e >> 1;
        int K = ((j >= 4) ? 16 : 0) + ((lane >= 16) ? 8 : 0) + ((j & 3) << 1) + (e & 1);
        float v = 0.0f;
        if      (K == 0)  v = x[(size_t)row * T_ALL + t];
        else if (K == 1)  v = cx;
        else if (K == 2)  v = cy;
        else if (K <= 19) v = env[((size_t)row * 17 + (K - 3)) * T_ALL + t];
        else if (K == 20) v = ar;
        o[e] = (bf16)v;
    }
}

__global__ void prep_bias(const float* __restrict__ ebih, const float* __restrict__ ebhh,
                          const float* __restrict__ ibih, const float* __restrict__ ibhh,
                          const float* __restrict__ dbih, const float* __restrict__ dbhh,
                          float* __restrict__ outb)
{
    int tid = blockIdx.x * blockDim.x + threadIdx.x;
    if (tid >= 2048) return;
    int i = tid & 511, w = tid >> 9;
    float v;
    if      (w == 0) v = ebih[i]       + ebhh[i];
    else if (w == 1) v = ebih[512 + i] + ebhh[512 + i];
    else if (w == 2) v = ibih[i]       + ibhh[i];
    else             v = dbih[i]       + dbhh[i];
    outb[tid] = v;
}

// ---------------------------------------------------------------------------
// Main kernel helpers
// ---------------------------------------------------------------------------
DEVINL float sigf(float x) { return 1.0f / (1.0f + __expf(-x)); }
DEVINL float tanh_f(float x) {
    float ax = fabsf(x);
    float e  = __expf(-2.0f * ax);
    float t  = (1.0f - e) / (1.0f + e);
    return x < 0.0f ? -t : t;
}

// Build a 16x32 bf16 A-fragment from row-major LDS (stride SW).
// Per lane: two contiguous 16B ds_load_b128s.
DEVINL v16bf ldsA(const bf16* base, int lane, int kOff)
{
    int row = lane & 15;
    int k0  = kOff + ((lane >> 4) << 3);   // upper half-wave: +8
    const bf16* p = base + row * SW + k0;
    v8bf lo = *(const v8bf*)p;
    v8bf hi = *(const v8bf*)(p + 16);
    return __builtin_shufflevector(lo, hi, 0,1,2,3,4,5,6,7,8,9,10,11,12,13,14,15);
}

// Store a 16x16 f32 D-tile (C/D VGPR layout) to row-major bf16 LDS.
DEVINL void stTile(bf16* base, int lane, int colBase, v8f acc)
{
    int hi8 = (lane >> 4) << 3;
    int col = colBase + (lane & 15);
    #pragma unroll
    for (int r = 0; r < 8; ++r)
        base[(r + hi8) * SW + col] = (bf16)acc[r];
}

DEVINL v8f wmma_bf(v16bf a, v16bf b, v8f c)
{
    return __builtin_amdgcn_wmma_f32_16x16x32_bf16(false, a, false, b, (short)0, c,
                                                   false, false);
}

// 4 tiles (nt = ntBase..ntBase+3) of an input projection: D = A @ B, bf16->LDS.
// All 4 WMMAs issue into independent accumulators before any conversion, so
// the WMMA->VALU hazard is paid once per group, not once per WMMA.
DEVINL void proj_stage(v16bf Ainp, const bf16* __restrict__ w, bf16* sdst,
                       int lane, int ntBase)
{
    v8f pacc[4];
    #pragma unroll
    for (int i = 0; i < 4; ++i) {
        v8f a;
        #pragma unroll
        for (int r = 0; r < 8; ++r) a[r] = 0.0f;
        pacc[i] = wmma_bf(Ainp,
            *(const v16bf*)(w + (size_t)(ntBase + i) * FRAG + lane * 16), a);
    }
    #pragma unroll
    for (int i = 0; i < 4; ++i)
        stTile(sdst, lane, (ntBase + i) * 16, pacc[i]);
}

// One LSTM layer, this wave computing column tiles t8 = t8base..t8base+3:
// gates = Ain@Wih.T + Ah@Whh.T + bias (bih+bhh combined), then
// c' = sig(f)*c + sig(i)*tanh(g);  h' = sig(o)*tanh(c')  -> s_hout (bf16)
// c is updated in-place: each (row,col) slot is touched only by its owner lane.
template<int KTIN>
DEVINL void lstm_layer(const v16bf* Ain, const v16bf* Ah,
                       const bf16* __restrict__ wih, const bf16* __restrict__ whh,
                       const float* __restrict__ bias,
                       float* s_c, bf16* s_hout, int lane, int t8base)
{
    const int col = lane & 15;
    const int hi8 = (lane >> 4) << 3;
    for (int ii = 0; ii < 4; ++ii) {
        const int t8 = t8base + ii;
        v8f acc[4];
        #pragma unroll
        for (int p = 0; p < 4; ++p) {         // i, f, g, o
            const int nt = p * 8 + t8;
            const float b = bias[nt * 16 + col];
            v8f a;
            #pragma unroll
            for (int r = 0; r < 8; ++r) a[r] = b;
            const bf16* wi = wih + (size_t)nt * KTIN * FRAG;
            #pragma unroll
            for (int kt = 0; kt < KTIN; ++kt)
                a = wmma_bf(Ain[kt], *(const v16bf*)(wi + kt * FRAG + lane * 16), a);
            const bf16* wh = whh + (size_t)nt * 4 * FRAG;
            #pragma unroll
            for (int kt = 0; kt < 4; ++kt)
                a = wmma_bf(Ah[kt], *(const v16bf*)(wh + kt * FRAG + lane * 16), a);
            acc[p] = a;
        }
        #pragma unroll
        for (int r = 0; r < 8; ++r) {
            const int ro = (r + hi8) * SWC + t8 * 16 + col;
            const int ho = (r + hi8) * SW  + t8 * 16 + col;
            float cold = s_c[ro];
            float iv = sigf(acc[0][r]);
            float fv = sigf(acc[1][r]);
            float gv = tanh_f(acc[2][r]);
            float ov = sigf(acc[3][r]);
            float cn = fv * cold + iv * gv;
            s_c[ro]    = cn;
            s_hout[ho] = (bf16)(ov * tanh_f(cn));
        }
    }
}

// ---------------------------------------------------------------------------
// Main persistent kernel: 2 wave32s per block share 16 batch rows.
// Wave w owns gate tiles t8 = 4w..4w+3 and output tiles nt = 4w..4w+3.
// h buffers ping-pong each layer -> only producer->consumer barriers needed.
// __launch_bounds__(64,8) pins VGPRs <= 256 logical: no vgpr-msb prefixes.
// ---------------------------------------------------------------------------
__global__ __launch_bounds__(64, 8) void lstm_persistent(
    const bf16* __restrict__ pInp,
    const bf16* __restrict__ wbase,     // all packed weights + combined biases
    const float* __restrict__ x,
    const float* __restrict__ mb1, const float* __restrict__ mb2,
    float* __restrict__ out)
{
    __shared__ __align__(16) bf16 s_h0a[16 * SW], s_h0b[16 * SW];
    __shared__ __align__(16) bf16 s_h1a[16 * SW], s_h1b[16 * SW];
    __shared__ __align__(16) bf16 s_proj[16 * SW];
    __shared__ __align__(16) bf16 s_z[16 * SW];
    __shared__ __align__(16) float s_c0[16 * SWC];
    __shared__ __align__(16) float s_c1[16 * SWC];
    __shared__ float s_ss[32];

    const int g    = blockIdx.x;
    const int tid  = threadIdx.x;
    const int lane = tid & 31;
    const int wv   = tid >> 5;          // 0 or 1
    const int col  = lane & 15;
    const int hi8  = (lane >> 4) << 3;
    const int tb   = wv * 4;            // this wave's tile base

    const bf16* wEncIn = wbase;
    const bf16* wih0   = wbase + (OFF_WIH0   - OFF_ENC_IN);
    const bf16* whh0   = wbase + (OFF_WHH0   - OFF_ENC_IN);
    const bf16* wih1   = wbase + (OFF_WIH1   - OFF_ENC_IN);
    const bf16* whh1   = wbase + (OFF_WHH1   - OFF_ENC_IN);
    const bf16* wDecIn = wbase + (OFF_DEC_IN - OFF_ENC_IN);
    const bf16* wInWih = wbase + (OFF_INWIH  - OFF_ENC_IN);
    const bf16* wInWhh = wbase + (OFF_INWHH  - OFF_ENC_IN);
    const bf16* wDWih  = wbase + (OFF_DWIH   - OFF_ENC_IN);
    const bf16* wDWhh  = wbase + (OFF_DWHH   - OFF_ENC_IN);
    const bf16* wM1    = wbase + (OFF_M1     - OFF_ENC_IN);
    const bf16* wM2    = wbase + (OFF_M2     - OFF_ENC_IN);
    const float* biasAll = (const float*)(wbase + (OFF_END - OFF_ENC_IN));
    const float* bEnc0 = biasAll;
    const float* bEnc1 = biasAll + 512;
    const float* bDec0 = biasAll + 1024;
    const float* bDec1 = biasAll + 1536;

    for (int i = tid; i < 16 * SW;  i += 64) {
        s_h0a[i] = (bf16)0.0f; s_h1a[i] = (bf16)0.0f;
    }
    for (int i = tid; i < 16 * SWC; i += 64) { s_c0[i] = 0.0f; s_c1[i] = 0.0f; }
    __syncthreads();

    const bf16* inpG = pInp + (size_t)g * T_ALL * 512;

    bf16* h0c = s_h0a; bf16* h0n = s_h0b;   // ping-pong h buffers
    bf16* h1c = s_h1a; bf16* h1n = s_h1b;

    v16bf Ax[8], Ah[4];

    // ---------------- encoder: 24 steps, 2-layer LSTM ----------------
    for (int t = 0; t < T_CTX; ++t) {
        v16bf Ainp = *(const v16bf*)(inpG + (size_t)t * 512 + lane * 16);
        __builtin_prefetch(inpG + (size_t)(t + 1) * 512 + lane * 16, 0, 3);
        proj_stage(Ainp, wEncIn, s_proj, lane, tb);          // proj = inp @ Wenc.T
        __syncthreads();                                     // proj visible
        #pragma unroll
        for (int kt = 0; kt < 4; ++kt) {
            Ax[kt] = ldsA(s_proj, lane, kt * 32);
            Ah[kt] = ldsA(h0c,    lane, kt * 32);
        }
        lstm_layer<4>(Ax, Ah, wih0, whh0, bEnc0, s_c0, h0n, lane, tb);
        __syncthreads();                                     // h0n visible
        #pragma unroll
        for (int kt = 0; kt < 4; ++kt) {
            Ax[kt] = ldsA(h0n, lane, kt * 32);
            Ah[kt] = ldsA(h1c, lane, kt * 32);
        }
        lstm_layer<4>(Ax, Ah, wih1, whh1, bEnc1, s_c1, h1n, lane, tb);
        __syncthreads();                                     // h1n visible
        bf16* tp;
        tp = h0c; h0c = h0n; h0n = tp;
        tp = h1c; h1c = h1n; h1n = tp;
    }

    // enc_state = h1 : hold A-fragments in registers for the whole decoder
    v16bf Aenc[4];
    #pragma unroll
    for (int kt = 0; kt < 4; ++kt) Aenc[kt] = ldsA(h1c, lane, kt * 32);

    float xc = x[(size_t)(g * 16 + col) * T_ALL + (T_CTX - 1)];

    // ---------------- decoder: 40 steps, 2-layer LSTM + MLP head -----
    for (int st = 0; st < HORIZ; ++st) {
        const int t = T_CTX + st;
        v16bf Ainp = *(const v16bf*)(inpG + (size_t)t * 512 + lane * 16);
        __builtin_prefetch(inpG + (size_t)(t + 1) * 512 + lane * 16, 0, 3);
        // splice recurrent xc into feature 0 (K=0 lives only in lanes 0..15, e=0)
        Ainp[0] = (lane < 16) ? (bf16)xc : Ainp[0];

        proj_stage(Ainp, wDecIn, s_proj, lane, tb);          // proj = inp @ Wdec.T
        __syncthreads();                                     // proj visible
        #pragma unroll
        for (int kt = 0; kt < 4; ++kt) {
            Ax[kt]     = ldsA(s_proj, lane, kt * 32);
            Ax[4 + kt] = Aenc[kt];            // concat [proj | enc_state], K=256
            Ah[kt]     = ldsA(h0c, lane, kt * 32);
        }
        lstm_layer<8>(Ax, Ah, wInWih, wInWhh, bDec0, s_c0, h0n, lane, tb);
        __syncthreads();                                     // h0n visible
        #pragma unroll
        for (int kt = 0; kt < 4; ++kt) {
            Ax[kt] = ldsA(h0n, lane, kt * 32);
            Ah[kt] = ldsA(h1c, lane, kt * 32);
        }
        lstm_layer<4>(Ax, Ah, wDWih, wDWhh, bDec1, s_c1, h1n, lane, tb);
        __syncthreads();                                     // h1n visible

        // MLP: zmid = leaky_relu([h1 | inp] @ W1.T + b1)  (K = 128 + 32pad)
        v16bf Am[5];
        #pragma unroll
        for (int kt = 0; kt < 4; ++kt) Am[kt] = ldsA(h1n, lane, kt * 32);
        Am[4] = Ainp;                          // inp (incl. xc) is K-tile 4
        #pragma unroll
        for (int i = 0; i < 4; ++i) {
            const int nt = tb + i;
            float b = mb1[nt * 16 + col];
            v8f a;
            #pragma unroll
            for (int r = 0; r < 8; ++r) a[r] = b;
            const bf16* w = wM1 + (size_t)nt * 5 * FRAG;
            #pragma unroll
            for (int kt = 0; kt < 5; ++kt)
                a = wmma_bf(Am[kt], *(const v16bf*)(w + kt * FRAG + lane * 16), a);
            #pragma unroll
            for (int r = 0; r < 8; ++r) a[r] = (a[r] > 0.0f) ? a[r] : 0.01f * a[r];
            stTile(s_z, lane, nt * 16, a);
        }
        __syncthreads();                                     // z visible

        // ss = zmid @ W2.T + b2  (2 valid output cols, padded to 16) — wave 0
        if (wv == 0) {
            v16bf Az[4];
            #pragma unroll
            for (int kt = 0; kt < 4; ++kt) Az[kt] = ldsA(s_z, lane, kt * 32);
            float b2 = (col < 2) ? mb2[col] : 0.0f;
            v8f ss;
            #pragma unroll
            for (int r = 0; r < 8; ++r) ss[r] = b2;
            #pragma unroll
            for (int kt = 0; kt < 4; ++kt)
                ss = wmma_bf(Az[kt], *(const v16bf*)(wM2 + kt * FRAG + lane * 16), ss);
            if (col < 2) {
                #pragma unroll
                for (int r = 0; r < 8; ++r) s_ss[(r + hi8) * 2 + col] = ss[r];
            }
        }
        __syncthreads();                                     // ss visible
        float s0 = s_ss[col * 2 + 0];
        float s1 = s_ss[col * 2 + 1];
        xc = xc + sigf(s0) - sigf(s1) * xc;    // xn = xc + source - sink
        if (wv == 0 && lane < 16)
            out[(size_t)(g * 16 + lane) * HORIZ + st] = xc;

        bf16* tp;
        tp = h0c; h0c = h0n; h0n = tp;
        tp = h1c; h1c = h1n; h1n = tp;
    }
}

// ---------------------------------------------------------------------------
extern "C" void kernel_launch(void* const* d_in, const int* in_sizes, int n_in,
                              void* d_out, int out_size, void* d_ws, size_t ws_size,
                              hipStream_t stream)
{
    const float* x      = (const float*)d_in[0];
    const float* coords = (const float*)d_in[1];
    const float* env    = (const float*)d_in[2];
    const float* areas  = (const float*)d_in[3];
    const float* WencIn = (const float*)d_in[4];
    const float* encWih = (const float*)d_in[5];
    const float* encWhh = (const float*)d_in[6];
    const float* encBih = (const float*)d_in[7];
    const float* encBhh = (const float*)d_in[8];
    const float* WdecIn = (const float*)d_in[9];
    const float* inWih  = (const float*)d_in[10];
    const float* inWhh  = (const float*)d_in[11];
    const float* inBih  = (const float*)d_in[12];
    const float* inBhh  = (const float*)d_in[13];
    const float* dWih   = (const float*)d_in[14];
    const float* dWhh   = (const float*)d_in[15];
    const float* dBih   = (const float*)d_in[16];
    const float* dBhh   = (const float*)d_in[17];
    const float* m1     = (const float*)d_in[18];
    const float* mb1    = (const float*)d_in[19];
    const float* m2     = (const float*)d_in[20];
    const float* mb2    = (const float*)d_in[21];
    float* out = (float*)d_out;

    bf16* base    = (bf16*)d_ws;                 // needs ~42.2 MB of workspace
    bf16* pInp    = base + OFF_INP;
    bf16* pEncIn  = base + OFF_ENC_IN;
    bf16* pWih0   = base + OFF_WIH0;
    bf16* pWhh0   = base + OFF_WHH0;
    bf16* pWih1   = base + OFF_WIH1;
    bf16* pWhh1   = base + OFF_WHH1;
    bf16* pDecIn  = base + OFF_DEC_IN;
    bf16* pInWih  = base + OFF_INWIH;
    bf16* pInWhh  = base + OFF_INWHH;
    bf16* pDWih   = base + OFF_DWIH;
    bf16* pDWhh   = base + OFF_DWHH;
    bf16* pM1     = base + OFF_M1;
    bf16* pM2     = base + OFF_M2;
    float* biasAll = (float*)(base + OFF_END);

    auto pg = [](int frags) { return dim3((unsigned)((frags * 32 + 255) / 256)); };

    pack_wt<<<pg(8),   256, 0, stream>>>(WencIn,             pEncIn, 128,  21,  8, 1);
    pack_wt<<<pg(128), 256, 0, stream>>>(encWih,             pWih0,  512, 128, 32, 4);
    pack_wt<<<pg(128), 256, 0, stream>>>(encWhh,             pWhh0,  512, 128, 32, 4);
    pack_wt<<<pg(128), 256, 0, stream>>>(encWih + 512 * 128, pWih1,  512, 128, 32, 4);
    pack_wt<<<pg(128), 256, 0, stream>>>(encWhh + 512 * 128, pWhh1,  512, 128, 32, 4);
    pack_wt<<<pg(8),   256, 0, stream>>>(WdecIn,             pDecIn, 128,  21,  8, 1);
    pack_wt<<<pg(256), 256, 0, stream>>>(inWih,              pInWih, 512, 256, 32, 8);
    pack_wt<<<pg(128), 256, 0, stream>>>(inWhh,              pInWhh, 512, 128, 32, 4);
    pack_wt<<<pg(128), 256, 0, stream>>>(dWih,               pDWih,  512, 128, 32, 4);
    pack_wt<<<pg(128), 256, 0, stream>>>(dWhh,               pDWhh,  512, 128, 32, 4);
    pack_wt<<<pg(40),  256, 0, stream>>>(m1,                 pM1,    128, 149,  8, 5);
    pack_wt<<<pg(4),   256, 0, stream>>>(m2,                 pM2,      2, 128,  1, 4);

    prep_bias<<<8, 256, 0, stream>>>(encBih, encBhh, inBih, inBhh, dBih, dBhh, biasAll);

    pack_inp<<<(NGROUP * T_ALL * 32 + 255) / 256, 256, 0, stream>>>(
        x, coords, env, areas, pInp);

    lstm_persistent<<<NGROUP, 64, 0, stream>>>(
        pInp, pEncIn /* weights base */, x, mb1, mb2, out);
}
